// CustomMixedDecoder_47175920779908
// MI455X (gfx1250) — compile-verified, matrix operand order checked by reading, not run on hardware
//
#include <hip/hip_runtime.h>
#include <stdint.h>

// Problem constants (from reference)
#define B_SAMPLES 1024
#define LATENT    32
#define FRAME     267
#define HID       512
#define NEXP      16
#define GATE_H    64
#define IN0       (LATENT + FRAME)   // 299
#define INTERSZ   (LATENT + HID)     // 544

typedef __attribute__((ext_vector_type(16))) __bf16 bf16x16;
typedef __attribute__((ext_vector_type(8)))  float  f32x8;

union FragU { uint4 u4[2]; uint32_t d[8]; bf16x16 v; };
union AccU  { f32x8 v; float f[8]; };

__device__ __forceinline__ uint32_t pack2_bf16(float a, float b) {
  uint32_t ua = __float_as_uint(a);
  uint32_t ub = __float_as_uint(b);
  ua = (ua + 0x7FFFu + ((ua >> 16) & 1u)) >> 16;   // round-to-nearest-even
  ub = (ub + 0x7FFFu + ((ub >> 16) & 1u)) >> 16;
  return ua | (ub << 16);
}

__device__ __forceinline__ float elu1(float x) {
  return x > 0.0f ? x : (__expf(x) - 1.0f);
}

// CDNA5 async global->LDS copy (tracked by ASYNCcnt), per-lane 16 bytes.
__device__ __forceinline__ void async_b128(uint32_t lds_byte, const uint32_t* g) {
  asm volatile("global_load_async_to_lds_b128 %0, %1, off"
               :: "v"(lds_byte), "v"(g) : "memory");
}
__device__ __forceinline__ void wait_async0() {
  asm volatile("s_wait_asynccnt 0" ::: "memory");
}

// ---------------------------------------------------------------------------
// Gate MLP: coef = softmax(elu(elu(x@gw1+gb1)@gw2+gb2)@gw3+gb3), x = [z|c]
// ---------------------------------------------------------------------------
__global__ void gate_kernel(const float* __restrict__ z, const float* __restrict__ c,
                            const float* __restrict__ gw1, const float* __restrict__ gb1,
                            const float* __restrict__ gw2, const float* __restrict__ gb2,
                            const float* __restrict__ gw3, const float* __restrict__ gb3,
                            float* __restrict__ coef) {
  extern __shared__ float sg[];
  float* W1 = sg;                          // 299*64
  float* B1 = W1 + IN0 * GATE_H;           // 64
  float* W2 = B1 + GATE_H;                 // 64*64
  float* B2 = W2 + GATE_H * GATE_H;        // 64
  float* W3 = B2 + GATE_H;                 // 64*16
  float* B3 = W3 + GATE_H * NEXP;          // 16

  const int tid = threadIdx.x;
  for (int i = tid; i < IN0 * GATE_H; i += 256) W1[i] = gw1[i];
  for (int i = tid; i < GATE_H * GATE_H; i += 256) W2[i] = gw2[i];
  for (int i = tid; i < GATE_H * NEXP; i += 256) W3[i] = gw3[i];
  for (int i = tid; i < GATE_H; i += 256) { B1[i] = gb1[i]; B2[i] = gb2[i]; }
  for (int i = tid; i < NEXP; i += 256) B3[i] = gb3[i];
  __syncthreads();

  const int b = blockIdx.x * 256 + tid;

  float h1[GATE_H];
  #pragma unroll
  for (int j = 0; j < GATE_H; ++j) h1[j] = B1[j];
  for (int i = 0; i < IN0; ++i) {
    float xi = (i < LATENT) ? z[b * LATENT + i] : c[b * FRAME + (i - LATENT)];
    #pragma unroll
    for (int j = 0; j < GATE_H; ++j) h1[j] = __fmaf_rn(xi, W1[i * GATE_H + j], h1[j]);
  }
  #pragma unroll
  for (int j = 0; j < GATE_H; ++j) h1[j] = elu1(h1[j]);

  float h2[GATE_H];
  #pragma unroll
  for (int j = 0; j < GATE_H; ++j) h2[j] = B2[j];
  for (int i = 0; i < GATE_H; ++i) {
    float xi = h1[i];
    #pragma unroll
    for (int j = 0; j < GATE_H; ++j) h2[j] = __fmaf_rn(xi, W2[i * GATE_H + j], h2[j]);
  }
  #pragma unroll
  for (int j = 0; j < GATE_H; ++j) h2[j] = elu1(h2[j]);

  float lg[NEXP];
  #pragma unroll
  for (int j = 0; j < NEXP; ++j) lg[j] = B3[j];
  for (int i = 0; i < GATE_H; ++i) {
    float xi = h2[i];
    #pragma unroll
    for (int j = 0; j < NEXP; ++j) lg[j] = __fmaf_rn(xi, W3[i * NEXP + j], lg[j]);
  }
  float m = lg[0];
  #pragma unroll
  for (int j = 1; j < NEXP; ++j) m = fmaxf(m, lg[j]);
  float s = 0.0f;
  #pragma unroll
  for (int j = 0; j < NEXP; ++j) { lg[j] = __expf(lg[j] - m); s += lg[j]; }
  float inv = 1.0f / s;
  #pragma unroll
  for (int j = 0; j < NEXP; ++j) coef[b * NEXP + j] = lg[j] * inv;
}

// ---------------------------------------------------------------------------
// Weight repack: w[e][k][n] fp32 -> packed bf16 chunks [e*KC+kc][n][16 dwords]
// dword (n, kd) packs (k = kc*32 + 2kd, k+1); K-tail and N-tail zero padded.
// Paid once per launch; makes the GEMM hot loop a flat aligned async copy.
// ---------------------------------------------------------------------------
__global__ void repack_w_kernel(const float* __restrict__ W, uint32_t* __restrict__ dst,
                                int K, int N, int KC, int Npad) {
  const int e  = blockIdx.y;
  const int kc = blockIdx.x;
  const int tid = threadIdx.x;
  const float* src = W + (size_t)e * K * N;
  uint32_t* out = dst + (size_t)(e * KC + kc) * Npad * 16;
  for (int n = tid; n < Npad; n += 256) {
    uint32_t row[16];
    #pragma unroll
    for (int kd = 0; kd < 16; ++kd) {
      int gk = kc * 32 + kd * 2;
      float f0 = 0.0f, f1 = 0.0f;
      if (n < N) {
        if (gk < K)     f0 = src[(size_t)gk * N + n];
        if (gk + 1 < K) f1 = src[(size_t)(gk + 1) * N + n];
      }
      row[kd] = pack2_bf16(f0, f1);
    }
    uint4* o4 = (uint4*)(out + (size_t)n * 16);
    o4[0] = *(uint4*)&row[0];
    o4[1] = *(uint4*)&row[4];
    o4[2] = *(uint4*)&row[8];
    o4[3] = *(uint4*)&row[12];
  }
}

// ---------------------------------------------------------------------------
// Mixed-expert layer: out[b,o] = act( sum_e coef[b,e]*(([z|prev]@W_e)[b,o]
//                                     + bias[e,o]) )
// WG = 128 threads (4 waves), 64x64 tile, wave tile 32x32 (2x2 WMMA bf16).
// A staged fully in LDS once; packed W chunks double-buffered via
// global_load_async_to_lds_b128. Per-expert f32 accumulators; blend+bias in
// the expert epilogue: accM += coef[b,e] * (accE + bias[e,o]).
// ---------------------------------------------------------------------------
template <int KC, int NPAD, int NREAL, int PREVC, bool ELU>
__global__ void moe_layer_kernel(const float* __restrict__ z,
                                 const float* __restrict__ prev,
                                 const uint32_t* __restrict__ pW,
                                 const float* __restrict__ bias,
                                 const float* __restrict__ coef,
                                 float* __restrict__ out) {
  constexpr int ASTRIDE = KC * 16 + 4;                 // A row pitch (dwords)
  constexpr int KD      = KC * 16;                     // A payload dwords/row
  constexpr int OFF_A   = 0;
  constexpr int OFF_C   = OFF_A + 64 * ASTRIDE;        // coef 64x17
  constexpr int OFF_BI  = OFF_C + 64 * 17;             // bias 16x64
  constexpr int OFF_B   = OFF_BI + 16 * 64;            // B dbl buffer 2x64x20
  constexpr int BSTRIDE = 20;                          // B row pitch (dwords)
  constexpr int T       = NEXP * KC;

  extern __shared__ uint32_t smem[];
  uint32_t* aT = smem + OFF_A;
  float*    cT = (float*)(smem + OFF_C);
  float*    bT = (float*)(smem + OFF_BI);

  const int tid = threadIdx.x;
  const int m0  = blockIdx.y * 64;
  const int n0  = blockIdx.x * 64;

  // Async B-chunk feed: flat aligned copy of one packed 64x16-dword chunk.
  const int brow = tid >> 1;             // 0..63
  const int h8   = (tid & 1) * 8;        // half-row dword offset
  const uint32_t* gB0 = pW + ((size_t)n0 + brow) * 16 + h8;
  auto issue_b_tile = [&](int t) {
    const uint32_t* gsrc = gB0 + (size_t)t * NPAD * 16;
    uint32_t dstb = (uint32_t)((OFF_B + (t & 1) * (64 * BSTRIDE) +
                                brow * BSTRIDE + h8) * 4);
    async_b128(dstb, gsrc);
    async_b128(dstb + 16, gsrc + 4);
  };
  issue_b_tile(0);

  // ---- Phase 1: stage A (bf16, zero-padded), coef, bias ----
  for (int row = 0; row < 64; ++row) {
    int gb = m0 + row;
    const float* zr = z + (size_t)gb * LATENT;
    const float* pr = prev + (size_t)gb * PREVC;
    for (int kd = tid; kd < KD; kd += 128) {
      int k0 = kd * 2, k1 = kd * 2 + 1;
      float f0 = (k0 < LATENT) ? zr[k0]
               : ((k0 - LATENT < PREVC) ? pr[k0 - LATENT] : 0.0f);
      float f1 = (k1 < LATENT) ? zr[k1]
               : ((k1 - LATENT < PREVC) ? pr[k1 - LATENT] : 0.0f);
      aT[row * ASTRIDE + kd] = pack2_bf16(f0, f1);
    }
  }
  for (int d = tid; d < 64 * NEXP; d += 128) {
    int row = d >> 4, e = d & 15;
    cT[row * 17 + e] = coef[(m0 + row) * NEXP + e];
  }
  for (int d = tid; d < NEXP * 64; d += 128) {
    int e = d >> 6, n = d & 63;
    bT[e * 64 + n] = (n0 + n < NREAL) ? bias[e * NREAL + (n0 + n)] : 0.0f;
  }
  wait_async0();
  __syncthreads();

  // ---- Wave/lane geometry ----
  const int lane = tid & 31;
  const int wv   = tid >> 5;
  const int wm   = (wv >> 1) * 32;
  const int wn   = (wv & 1) * 32;
  const int half = lane >> 4;
  const int idx  = lane & 15;
  const int q    = half * 4;             // A/B fragment dword base (0 or 4)

  AccU accM[2][2];
  #pragma unroll
  for (int fm = 0; fm < 2; ++fm)
    #pragma unroll
    for (int fn = 0; fn < 2; ++fn)
      #pragma unroll
      for (int r = 0; r < 8; ++r) accM[fm][fn].f[r] = 0.0f;

  // ---- Main loop: experts outer, K-chunks inner ----
  for (int e = 0; e < NEXP; ++e) {
    AccU accE[2][2];
    #pragma unroll
    for (int fm = 0; fm < 2; ++fm)
      #pragma unroll
      for (int fn = 0; fn < 2; ++fn)
        #pragma unroll
        for (int r = 0; r < 8; ++r) accE[fm][fn].f[r] = 0.0f;

    for (int kc = 0; kc < KC; ++kc) {
      const int t = e * KC + kc;
      if (t + 1 < T) issue_b_tile(t + 1);

      const uint32_t* bufc = smem + OFF_B + (t & 1) * (64 * BSTRIDE);

      FragU afrag[2], bfrag[2];
      #pragma unroll
      for (int fm = 0; fm < 2; ++fm) {
        const uint32_t* p = aT + (wm + fm * 16 + idx) * ASTRIDE + kc * 16 + q;
        afrag[fm].u4[0] = *(const uint4*)(p);
        afrag[fm].u4[1] = *(const uint4*)(p + 8);
      }
      #pragma unroll
      for (int fn = 0; fn < 2; ++fn) {
        const uint32_t* p = bufc + (wn + fn * 16 + idx) * BSTRIDE + q;
        bfrag[fn].u4[0] = *(const uint4*)(p);
        bfrag[fn].u4[1] = *(const uint4*)(p + 8);
      }
      #pragma unroll
      for (int fm = 0; fm < 2; ++fm)
        #pragma unroll
        for (int fn = 0; fn < 2; ++fn)
          accE[fm][fn].v = __builtin_amdgcn_wmma_f32_16x16x32_bf16(
              false, afrag[fm].v, false, bfrag[fn].v,
              (short)0, accE[fm][fn].v, false, false);

      wait_async0();
      __syncthreads();
    }

    // ---- Expert epilogue: accM += coef[b,e] * (accE + bias[e,o]) ----
    float cf[2][8];
    #pragma unroll
    for (int fm = 0; fm < 2; ++fm)
      #pragma unroll
      for (int r = 0; r < 8; ++r)
        cf[fm][r] = cT[(wm + fm * 16 + r + 8 * half) * 17 + e];
    float bv[2];
    #pragma unroll
    for (int fn = 0; fn < 2; ++fn)
      bv[fn] = bT[e * 64 + wn + fn * 16 + idx];
    #pragma unroll
    for (int fm = 0; fm < 2; ++fm)
      #pragma unroll
      for (int fn = 0; fn < 2; ++fn)
        #pragma unroll
        for (int r = 0; r < 8; ++r)
          accM[fm][fn].f[r] = __fmaf_rn(cf[fm][r], accE[fm][fn].f[r] + bv[fn],
                                        accM[fm][fn].f[r]);
  }

  // ---- Final epilogue: optional ELU + store ----
  #pragma unroll
  for (int fm = 0; fm < 2; ++fm)
    #pragma unroll
    for (int fn = 0; fn < 2; ++fn)
      #pragma unroll
      for (int r = 0; r < 8; ++r) {
        int gb = m0 + wm + fm * 16 + r + 8 * half;
        int go = n0 + wn + fn * 16 + idx;
        if (go < NREAL) {
          float v = accM[fm][fn].f[r];
          if (ELU) v = elu1(v);
          out[(size_t)gb * NREAL + go] = v;
        }
      }
}

// ---------------------------------------------------------------------------
// Host-side launch
// ---------------------------------------------------------------------------
static inline size_t layer_lds_bytes(int KC) {
  int astride = KC * 16 + 4;
  return (size_t)(64 * astride + 64 * 17 + 16 * 64 + 2 * 64 * 20) * 4;
}

extern "C" void kernel_launch(void* const* d_in, const int* in_sizes, int n_in,
                              void* d_out, int out_size, void* d_ws, size_t ws_size,
                              hipStream_t stream) {
  const float* z   = (const float*)d_in[0];
  const float* c   = (const float*)d_in[1];
  const float* gw1 = (const float*)d_in[2];
  const float* gb1 = (const float*)d_in[3];
  const float* gw2 = (const float*)d_in[4];
  const float* gb2 = (const float*)d_in[5];
  const float* gw3 = (const float*)d_in[6];
  const float* gb3 = (const float*)d_in[7];
  const float* w0  = (const float*)d_in[8];
  const float* b0  = (const float*)d_in[9];
  const float* w1  = (const float*)d_in[10];
  const float* b1  = (const float*)d_in[11];
  const float* w2  = (const float*)d_in[12];
  const float* b2  = (const float*)d_in[13];

  // Workspace layout (dwords)
  float*    coef = (float*)d_ws;                                   // 1024*16
  float*    act0 = coef + (size_t)B_SAMPLES * NEXP;                // 1024*512
  float*    act1 = act0 + (size_t)B_SAMPLES * HID;                 // 1024*512
  uint32_t* pw0  = (uint32_t*)(act1 + (size_t)B_SAMPLES * HID);    // 16*10*512*16
  uint32_t* pw1  = pw0 + (size_t)NEXP * 10 * 512 * 16;             // 16*17*512*16
  uint32_t* pw2  = pw1 + (size_t)NEXP * 17 * 512 * 16;             // 16*17*320*16
  float*    outp = (float*)d_out;                                  // 1024*267

  // Weight repacks (fp32 -> packed bf16 chunks) + gate MLP
  repack_w_kernel<<<dim3(10, NEXP), dim3(256), 0, stream>>>(w0, pw0, IN0, HID, 10, 512);
  repack_w_kernel<<<dim3(17, NEXP), dim3(256), 0, stream>>>(w1, pw1, INTERSZ, HID, 17, 512);
  repack_w_kernel<<<dim3(17, NEXP), dim3(256), 0, stream>>>(w2, pw2, INTERSZ, FRAME, 17, 320);

  size_t gate_lds = (size_t)(IN0 * GATE_H + GATE_H + GATE_H * GATE_H + GATE_H +
                             GATE_H * NEXP + NEXP) * 4;
  gate_kernel<<<dim3(B_SAMPLES / 256), dim3(256), gate_lds, stream>>>(
      z, c, gw1, gb1, gw2, gb2, gw3, gb3, coef);

  // Layer 0: [z|c] (K=299 -> KC=10) -> 512, ELU
  moe_layer_kernel<10, 512, 512, FRAME, true>
      <<<dim3(512 / 64, B_SAMPLES / 64), dim3(128), layer_lds_bytes(10), stream>>>(
          z, c, pw0, b0, coef, act0);
  // Layer 1: [z|act0] (K=544 -> KC=17) -> 512, ELU
  moe_layer_kernel<17, 512, 512, HID, true>
      <<<dim3(512 / 64, B_SAMPLES / 64), dim3(128), layer_lds_bytes(17), stream>>>(
          z, act0, pw1, b1, coef, act1);
  // Layer 2: [z|act1] (K=544 -> KC=17) -> 267 (NPAD=320), no activation
  moe_layer_kernel<17, 320, FRAME, HID, false>
      <<<dim3(320 / 64, B_SAMPLES / 64), dim3(128), layer_lds_bytes(17), stream>>>(
          z, act1, pw2, b2, coef, outp);
}
